// LSTM_model_27333171872212
// MI455X (gfx1250) — matrix operand order repaired and checked by reading, not compiled, not run on hardware
//
#include <hip/hip_runtime.h>
#include <hip/hip_bf16.h>

// ---------------------------------------------------------------------------
// LSTM on MI455X (gfx1250):
//   phase 1: xW = x @ W + hid_bias          (bf16 WMMA GEMM, 32768x2048x512,
//            B-tile staged to LDS with async global->LDS copies)
//   phase 2: 1024-step recurrence           (persistent 16-WG cluster kernel,
//            128KB U slice loaded to LDS via TDM tensor_load_to_lds,
//            h broadcast staged via async-to-LDS each step,
//            s_cluster_barrier + atomic grid sync per step)
//   phase 3: out = hidden @ V + out_bias    (bf16 WMMA GEMM, 32768x512x512)
// ---------------------------------------------------------------------------

typedef __bf16 bf16_t;
typedef __attribute__((ext_vector_type(16))) __bf16 v16bf;
typedef __attribute__((ext_vector_type(8)))  __bf16 v8bf;
typedef __attribute__((ext_vector_type(8)))  float  v8f;
typedef __attribute__((ext_vector_type(4)))  unsigned int v4u;
typedef __attribute__((ext_vector_type(4)))  int v4i;
typedef __attribute__((ext_vector_type(8)))  int v8i;

#define LSTM_B  32
#define LSTM_S  1024
#define LSTM_NI 512
#define LSTM_NH 512
#define LSTM_NO 512
#define LSTM_G  (4 * LSTM_NH)   // 2048
#define NWG     16              // workgroups in the recurrent "cluster"

#if __has_builtin(__builtin_amdgcn_global_load_async_to_lds_b128)
#define HAVE_ASYNC_LDS 1
#endif
#if __has_builtin(__builtin_amdgcn_tensor_load_to_lds)
#define HAVE_TDM 1
#endif

// ---- async global->LDS 16B copy (ASYNCcnt-tracked), with sync fallback ----
// Prototype (from hipcc diagnostics): param0 = int4 in AS(1) [global],
// param1 = int4 in AS(3) [LDS], then (int offset, int cpol).
static __device__ __forceinline__ void async_copy16(const bf16_t* g, bf16_t* l) {
#ifdef HAVE_ASYNC_LDS
  __builtin_amdgcn_global_load_async_to_lds_b128(
      (__attribute__((address_space(1))) v4i*)g,
      (__attribute__((address_space(3))) v4i*)l, 0, 0);
#else
  *(v8bf*)l = *(const v8bf*)g;
#endif
}

static __device__ __forceinline__ void async_wait0() {
#if defined(HAVE_ASYNC_LDS) && __has_builtin(__builtin_amdgcn_s_wait_asynccnt)
  __builtin_amdgcn_s_wait_asynccnt(0);
#endif
}

// ---- TDM: DMA `rows` contiguous rows of rowDwords*4 bytes into LDS ---------
// 2D descriptor per CDNA5 ISA ch.8: data_size=4B, tile = rowDwords x rows.
static __device__ __forceinline__ void tdm_load_rows(const void* gptr,
                                                     unsigned ldsAddr,
                                                     unsigned rows,
                                                     unsigned rowDwords) {
#ifdef HAVE_TDM
  unsigned long long ga = (unsigned long long)gptr;
  v4u g0;
  g0[0] = 1u;                                             // count=1 (user D#)
  g0[1] = ldsAddr;                                        // lds_addr
  g0[2] = (unsigned)(ga & 0xFFFFFFFFu);                   // global_addr lo
  g0[3] = (unsigned)((ga >> 32) & 0x01FFFFFFu) | (2u << 30);  // hi | type=2
  v8i g1;
  g1[0] = (int)(2u << 16);                                // data_size=2 -> 4B
  g1[1] = (int)((rowDwords & 0xFFFFu) << 16);             // tensor_dim0[15:0]
  g1[2] = (int)(((rowDwords >> 16) & 0xFFFFu) |
                ((2048u & 0xFFFFu) << 16));               // dim0 hi | dim1 lo
  g1[3] = (int)((rowDwords & 0xFFFFu) << 16);             // tile_dim0
  g1[4] = (int)(rows & 0xFFFFu);                          // tile_dim1 (2D)
  g1[5] = (int)rowDwords;                                 // dim0_stride lo
  g1[6] = 0;
  g1[7] = 0;
  v4i z = {0, 0, 0, 0};
#if __clang_major__ >= 23
  v8i z8 = {0, 0, 0, 0, 0, 0, 0, 0};
  __builtin_amdgcn_tensor_load_to_lds(g0, g1, z, z, z8, 0);
#else
  __builtin_amdgcn_tensor_load_to_lds(g0, g1, z, z, 0);
#endif
#endif
}

static __device__ __forceinline__ void tdm_wait0() {
#if defined(HAVE_TDM) && __has_builtin(__builtin_amdgcn_s_wait_tensorcnt)
  __builtin_amdgcn_s_wait_tensorcnt(0);
#endif
}

static __device__ __forceinline__ v16bf make_v16(v8bf lo, v8bf hi) {
  v16bf r;
#pragma unroll
  for (int i = 0; i < 8; ++i) { r[i] = lo[i]; r[8 + i] = hi[i]; }
  return r;
}

// A fragment (16x32 bf16, row-major source), per ISA 7.12.2:
// lanes 0-15: K = {0..7, 16..23}; lanes 16-31: K = {8..15, 24..31}
static __device__ __forceinline__ v16bf load_a_frag(const bf16_t* rowbase, int kb, int lane) {
  const int kb2 = (lane < 16) ? 0 : 8;
  v8bf lo = *(const v8bf*)(rowbase + kb + kb2);
  v8bf hi = *(const v8bf*)(rowbase + kb + 16 + kb2);
  return make_v16(lo, hi);
}

// B fragment (32x16 bf16) from N-major (pre-transposed) storage:
// lanes 0-15 hold K={0..7,16..23} of column n, lanes 16-31 hold K={8..15,24..31}
static __device__ __forceinline__ v16bf load_b_frag(const bf16_t* colbase, int kb, int lane) {
  const int kh = (lane < 16) ? 0 : 8;
  v8bf lo = *(const v8bf*)(colbase + kb + kh);
  v8bf hi = *(const v8bf*)(colbase + kb + 16 + kh);
  return make_v16(lo, hi);
}

static __device__ __forceinline__ v8f wmma_bf16(v16bf a, v16bf b, v8f c) {
  return __builtin_amdgcn_wmma_f32_16x16x32_bf16(false, a, false, b, (short)0, c,
                                                 false, false);
}

static __device__ __forceinline__ float sigmoidf_fast(float x) {
  return 1.0f / (1.0f + __expf(-x));
}

// ---------------------------------------------------------------------------
// bf16 WMMA GEMM:  C[M,N] (f32) = A[M,K] bf16 row-major * Bt[N,K] bf16 N-major
//                  + bias[N]
// The 64-column B tile (64KB) is staged once into LDS with async-to-LDS and
// shared by all 8 waves; A fragments stream from global (L2).
// mode 0 (phase 1): A row of logical m=(t*B+b) is x row (b*S+t); C row = m.
// mode 1 (phase 3): A row = m; C row of logical m=(s*B+b) is out row (b*S+s).
// ---------------------------------------------------------------------------
__global__ __launch_bounds__(256)
void gemm_bf16_wmma(const bf16_t* __restrict__ A, const bf16_t* __restrict__ Bt,
                    const float* __restrict__ bias, float* __restrict__ C,
                    int M, int N, int K, int S, int Bsz, int mode) {
  __shared__ bf16_t Bs[64 * 512];   // 64 KB: [n_local][k]

  const int tid   = threadIdx.x;
  const int lane  = tid & 31;
  const int wave  = tid >> 5;
  const int Mbase = blockIdx.y * 128 + wave * 16;
  const int Nbase = blockIdx.x * 64;

  // Stage Bt rows [Nbase, Nbase+64) x K into LDS (contiguous 64*K*2 bytes).
  {
    const bf16_t* bsrc = Bt + (size_t)Nbase * (size_t)K;
    const int chunks = (64 * 512) / 8;           // 16B chunks
    for (int c = tid; c < chunks; c += 256)
      async_copy16(bsrc + (size_t)c * 8, &Bs[c * 8]);
    async_wait0();
  }
  __syncthreads();

  const int m = Mbase + (lane & 15);
  const size_t arow = (mode == 0)
                          ? ((size_t)(m % Bsz) * (size_t)S + (size_t)(m / Bsz))
                          : (size_t)m;
  const bf16_t* arowp = A + arow * (size_t)K;

  v8f acc[4] = {};
  for (int kb = 0; kb < K; kb += 32) {
    __builtin_prefetch(arowp + kb + 128, 0, 1);  // global_prefetch_b8
    v16bf a = load_a_frag(arowp, kb, lane);
#pragma unroll
    for (int j = 0; j < 4; ++j) {
      const bf16_t* colp = &Bs[(j * 16 + (lane & 15)) * 512];
      v16bf b = load_b_frag(colp, kb, lane);
      acc[j] = wmma_bf16(a, b, acc[j]);
    }
  }

#pragma unroll
  for (int j = 0; j < 4; ++j) {
    const int col = Nbase + j * 16 + (lane & 15);
    const float bv = bias[col];
#pragma unroll
    for (int r = 0; r < 8; ++r) {
      const int mm = Mbase + r + ((lane < 16) ? 0 : 8);
      const size_t crow = (mode == 0)
                              ? (size_t)mm
                              : ((size_t)(mm % Bsz) * (size_t)S + (size_t)(mm / Bsz));
      C[crow * (size_t)N + col] = acc[j][r] + bv;
    }
  }
}

// ---------------------------------------------------------------------------
// Persistent recurrent kernel: 16 WGs x 256 threads (8 wave32 each).
// WG j owns h/c columns [j*32, j*32+32); its 4x(512x32) slice of U lives in
// 128KB of LDS (loaded once by TDM). Each step: h (32KB) is staged to LDS
// with async-to-LDS, then each wave runs 2x16 v_wmma_f32_16x16x32_bf16
// entirely out of LDS.
// ---------------------------------------------------------------------------
__global__ __launch_bounds__(256)
void lstm_scan(const float* __restrict__ xW, const bf16_t* __restrict__ Ut,
               bf16_t* __restrict__ h0buf, bf16_t* __restrict__ h1buf,
               bf16_t* __restrict__ hidden, unsigned int* __restrict__ sync,
               int S) {
  __shared__ bf16_t Uslice[4 * 32 * 512];   // 128 KB: [g][n][k], N-major
  __shared__ bf16_t hs[32 * 512];           // 32 KB: h staging, row-major
  __shared__ float  gbuf[4][32][32];        // activated gate tiles
  __shared__ float  cbuf[32 * 32];          // cell state slice

  const int j    = blockIdx.x;       // 0..15: which 32-column slice we own
  const int tid  = threadIdx.x;
  const int lane = tid & 31;
  const int wave = tid >> 5;

  // One-time U slice load: 4 TDM DMAs of 32 rows x 1KB each.
#ifdef HAVE_TDM
  if (wave == 0) {
    const unsigned ldsbase = (unsigned)(unsigned long long)(void*)&Uslice[0];
    for (int g = 0; g < 4; ++g)
      tdm_load_rows(Ut + (size_t)(g * 512 + j * 32) * 512,
                    ldsbase + (unsigned)g * 32u * 1024u, 32u, 256u);
    tdm_wait0();
  }
#else
  for (int c = tid; c < (128 * 512) / 8; c += 256) {   // 8192 16B chunks
    const int row = c >> 6;
    const int off = (c & 63) * 8;
    const int g = row >> 5, n = row & 31;
    const v8bf* src =
        (const v8bf*)(Ut + ((size_t)(g * 512 + j * 32 + n)) * 512 + off);
    *(v8bf*)(&Uslice[row * 512 + off]) = *src;
  }
#endif
  for (int c = tid; c < 32 * 32; c += 256) cbuf[c] = 0.0f;
  __syncthreads();

  const int g  = wave >> 1;   // gate: 0=i, 1=f, 2=g, 3=o
  const int nt = wave & 1;    // which 16-column half of our 32 columns
  const bf16_t* bbase = &Uslice[(g * 32 + nt * 16 + (lane & 15)) * 512];
  const bf16_t* a0base = &hs[(lane & 15) * 512];         // rows 0..15
  const bf16_t* a1base = &hs[(16 + (lane & 15)) * 512];  // rows 16..31

  for (int t = 0; t < S; ++t) {
    const bf16_t* hin  = (t & 1) ? h1buf : h0buf;
    bf16_t*       hout = (t & 1) ? h0buf : h1buf;

    // Stage h[32][512] (bf16, 32KB) into LDS via async copies.
    for (int c = tid; c < (32 * 512) / 8; c += 256)
      async_copy16(hin + (size_t)c * 8, &hs[c * 8]);
    async_wait0();
    __syncthreads();

    v8f acc0 = {}, acc1 = {};
#pragma unroll 4
    for (int kb = 0; kb < 512; kb += 32) {
      v16bf b  = load_b_frag(bbase, kb, lane);
      v16bf a0 = load_a_frag(a0base, kb, lane);
      v16bf a1 = load_a_frag(a1base, kb, lane);
      acc0 = wmma_bf16(a0, b, acc0);
      acc1 = wmma_bf16(a1, b, acc1);
    }

    // gates = acc + xW[t]; activate; exchange through LDS
    const float* xwrow = xW + (size_t)t * LSTM_B * LSTM_G;
    const int nl   = nt * 16 + (lane & 15);          // 0..31 within our slice
    const int colg = g * 512 + j * 32 + nl;          // column in [0,2048)
#pragma unroll
    for (int r = 0; r < 8; ++r) {
      const int m0 = r + ((lane < 16) ? 0 : 8);
      float v0 = acc0[r] + xwrow[(size_t)m0 * LSTM_G + colg];
      float v1 = acc1[r] + xwrow[(size_t)(m0 + 16) * LSTM_G + colg];
      if (g == 2) { v0 = tanhf(v0); v1 = tanhf(v1); }
      else        { v0 = sigmoidf_fast(v0); v1 = sigmoidf_fast(v1); }
      gbuf[g][m0][nl]      = v0;
      gbuf[g][m0 + 16][nl] = v1;
    }
    __syncthreads();

    // c/h update on our 32x32 slice (4 elements per thread)
#pragma unroll
    for (int q = 0; q < 4; ++q) {
      const int idx = tid + q * 256;
      const int m = idx >> 5, n = idx & 31;
      const float iv = gbuf[0][m][n];
      const float fv = gbuf[1][m][n];
      const float gv = gbuf[2][m][n];
      const float ov = gbuf[3][m][n];
      const float c = fv * cbuf[idx] + iv * gv;
      cbuf[idx] = c;
      const float h = ov * tanhf(c);
      const bf16_t hb = (bf16_t)h;
      hout[(size_t)m * LSTM_NH + j * 32 + n] = hb;
      hidden[((size_t)t * LSTM_B + m) * LSTM_NH + j * 32 + n] = hb;
    }
    __threadfence();        // release h stores to device scope (L2)
    __syncthreads();

    // Cross-WG step barrier: cluster barrier (fast path if cluster-launched;
    // architecturally a NOP when ClusterID==0) + atomic counter (plain launch).
    if (tid == 0)
      __hip_atomic_fetch_add(sync, 1u, __ATOMIC_RELEASE, __HIP_MEMORY_SCOPE_AGENT);
    __builtin_amdgcn_s_cluster_barrier();
    if (tid == 0) {
      const unsigned target = (unsigned)NWG * (unsigned)(t + 1);
      while (__hip_atomic_load(sync, __ATOMIC_ACQUIRE, __HIP_MEMORY_SCOPE_AGENT) < target)
        __builtin_amdgcn_s_sleep(1);
    }
    __syncthreads();
  }
}

// ---------------------------------------------------------------------------
// Prep kernels
// ---------------------------------------------------------------------------
__global__ void k_f32_to_bf16(const float* __restrict__ in, bf16_t* __restrict__ out,
                              size_t n) {
  size_t i = (size_t)blockIdx.x * blockDim.x + threadIdx.x;
  if (i < n) out[i] = (bf16_t)in[i];
}

// in [K][N] f32 -> out [N][K] bf16
__global__ void k_transpose_bf16(const float* __restrict__ in, bf16_t* __restrict__ out,
                                 int K, int N) {
  size_t i = (size_t)blockIdx.x * blockDim.x + threadIdx.x;
  if (i < (size_t)K * (size_t)N) {
    const int k = (int)(i / (size_t)N);
    const int n = (int)(i % (size_t)N);
    out[(size_t)n * (size_t)K + k] = (bf16_t)in[i];
  }
}

__global__ void k_init_state(bf16_t* __restrict__ h0, bf16_t* __restrict__ h1,
                             unsigned int* __restrict__ sync, int n) {
  int i = blockIdx.x * blockDim.x + threadIdx.x;
  if (i < n) { h0[i] = (bf16_t)0.0f; h1[i] = (bf16_t)0.0f; }
  if (i == 0) *sync = 0u;
}

// ---------------------------------------------------------------------------
extern "C" void kernel_launch(void* const* d_in, const int* in_sizes, int n_in,
                              void* d_out, int out_size, void* d_ws, size_t ws_size,
                              hipStream_t stream) {
  const float* x        = (const float*)d_in[0];  // [B,S,NI]
  const float* W        = (const float*)d_in[1];  // [NI,4H]
  const float* U        = (const float*)d_in[2];  // [NH,4H]
  const float* hid_bias = (const float*)d_in[3];  // [4H]
  const float* V        = (const float*)d_in[4];  // [NH,NO]
  const float* out_bias = (const float*)d_in[5];  // [NO]
  float* out = (float*)d_out;                     // [B,S,NO]

  char* ws = (char*)d_ws;
  size_t off = 0;
  auto alloc = [&](size_t bytes) -> char* {
    char* p = ws + off;
    off = (off + bytes + 255) & ~(size_t)255;
    return p;
  };

  const size_t nx   = (size_t)LSTM_B * LSTM_S * LSTM_NI;   // 16.7M
  const size_t nhid = (size_t)LSTM_S * LSTM_B * LSTM_NH;   // 16.7M

  float*  xW     = (float*)alloc((size_t)LSTM_S * LSTM_B * LSTM_G * sizeof(float)); // 256MB
  bf16_t* x_bf   = (bf16_t*)alloc(nx * sizeof(bf16_t));                             // 32MB
  bf16_t* Wt     = (bf16_t*)alloc((size_t)LSTM_G * LSTM_NI * sizeof(bf16_t));       // 2MB
  bf16_t* Ut     = (bf16_t*)alloc((size_t)LSTM_G * LSTM_NH * sizeof(bf16_t));       // 2MB
  bf16_t* Vt     = (bf16_t*)alloc((size_t)LSTM_NO * LSTM_NH * sizeof(bf16_t));      // 0.5MB
  bf16_t* hidden = (bf16_t*)alloc(nhid * sizeof(bf16_t));                           // 32MB
  bf16_t* h0     = (bf16_t*)alloc((size_t)LSTM_B * LSTM_NH * sizeof(bf16_t));
  bf16_t* h1     = (bf16_t*)alloc((size_t)LSTM_B * LSTM_NH * sizeof(bf16_t));
  unsigned int* sync = (unsigned int*)alloc(256);

  // 1) precision conversion / layout transforms
  k_f32_to_bf16<<<dim3((unsigned)((nx + 255) / 256)), dim3(256), 0, stream>>>(x, x_bf, nx);
  {
    const size_t n = (size_t)LSTM_NI * LSTM_G;
    k_transpose_bf16<<<dim3((unsigned)((n + 255) / 256)), dim3(256), 0, stream>>>(
        W, Wt, LSTM_NI, LSTM_G);
  }
  {
    const size_t n = (size_t)LSTM_NH * LSTM_G;
    k_transpose_bf16<<<dim3((unsigned)((n + 255) / 256)), dim3(256), 0, stream>>>(
        U, Ut, LSTM_NH, LSTM_G);
  }
  {
    const size_t n = (size_t)LSTM_NH * LSTM_NO;
    k_transpose_bf16<<<dim3((unsigned)((n + 255) / 256)), dim3(256), 0, stream>>>(
        V, Vt, LSTM_NH, LSTM_NO);
  }
  k_init_state<<<dim3(64), dim3(256), 0, stream>>>(h0, h1, sync, LSTM_B * LSTM_NH);

  // 2) phase 1: xW[S,B,4H] = x @ W + hid_bias   (M=32768, N=2048, K=512)
  gemm_bf16_wmma<<<dim3(LSTM_G / 64, (LSTM_B * LSTM_S) / 128), dim3(256), 0, stream>>>(
      x_bf, Wt, hid_bias, xW, LSTM_B * LSTM_S, LSTM_G, LSTM_NI, LSTM_S, LSTM_B,
      /*mode=*/0);

  // 3) phase 2: recurrence (persistent, 16 WGs)
  lstm_scan<<<dim3(NWG), dim3(256), 0, stream>>>(xW, Ut, h0, h1, hidden, sync, LSTM_S);

  // 4) phase 3: out[B,S,NO] = hidden @ V + out_bias  (M=32768, N=512, K=512)
  gemm_bf16_wmma<<<dim3(LSTM_NO / 64, (LSTM_B * LSTM_S) / 128), dim3(256), 0, stream>>>(
      hidden, Vt, out_bias, out, LSTM_B * LSTM_S, LSTM_NO, LSTM_NH, LSTM_S, LSTM_B,
      /*mode=*/1);
}
